// WindowAttention_34462817583439
// MI455X (gfx1250) — matrix operand order repaired and checked by reading, not compiled, not run on hardware
//
#include <hip/hip_runtime.h>
#include <math.h>

typedef __attribute__((ext_vector_type(16))) _Float16 v16h;
typedef __attribute__((ext_vector_type(8)))  _Float16 v8h;
typedef __attribute__((ext_vector_type(8)))  float    v8f;

#define NTOK   1024
#define HEADS  64
#define QKVLD  192
#define SCALE  0.125f            // 64^-0.5
#define THR    0.1f
#define LOGIT_THR (-2.1972246f)  // ln(0.1/0.9): sigmoid(v)>0.1 <=> v>LOGIT_THR
#define KM_BASE 5
#define KM_MAX  20

// ---------------------------------------------------------------------------
// Async global -> LDS copy (CDNA5), one b32 per lane.  Tracked by ASYNCcnt.
// ---------------------------------------------------------------------------
__device__ __forceinline__ void wa_async_g2l_b32(const void* g, void* l)
{
    asm volatile("global_load_async_to_lds_b32 %0, %1, off"
                 :
                 : "v"((unsigned)(unsigned long long)l), "v"(g)
                 : "memory");
}
__device__ __forceinline__ void wa_wait_async()
{
    asm volatile("s_wait_asynccnt 0x0" ::: "memory");
}

// ---------------------------------------------------------------------------
// Weight pre-pack: W[K,N] f32 -> Wp[(K/32)][N][32] f16 (B-fragment layout):
// each column's 32 k-values contiguous, so a lane's B fragment = 16
// contiguous halfs (one v16h / 2x b128 load).  Done once, reused by all tiles.
// ---------------------------------------------------------------------------
__global__ void wa_pack_w(const float* __restrict__ W, _Float16* __restrict__ Wp,
                          int K, int N)
{
    const int i = blockIdx.x * blockDim.x + threadIdx.x;
    if (i >= K * N) return;
    const int k = i / N, j = i - k * N;
    Wp[((size_t)(k >> 5) * N + j) * 32 + (k & 31)] = (_Float16)W[i];
}

__global__ void wa_cvt16(const float* __restrict__ s, _Float16* __restrict__ d, int n)
{
    const int i = blockIdx.x * blockDim.x + threadIdx.x;
    if (i < n) d[i] = (_Float16)s[i];
}

// ---------------------------------------------------------------------------
// f16-WMMA GEMM: C = A16[M,K] @ Wp + bias.  One wave per PAIR of adjacent
// 16x16 tiles (shared A fragment, two independent accumulators -> the
// unrolled stream interleaves acc0/acc1 WMMAs with no D->C hazard bubbles).
// A fragment: 2x contiguous v8h loads.  B fragment: 1x v16h load per tile.
// MODE 0: plain              MODE 1: xs = aux * (v > logit(0.1))
// MODE 2: v += aux           MODE 3: out = early ? aux : v
// WF32 / W16 select f32 and/or row-major-f16 outputs.
// ---------------------------------------------------------------------------
template<int K, int MODE, bool WF32, bool W16>
__global__ void wa_wmma_gemm(const _Float16* __restrict__ A16,
                             const _Float16* __restrict__ Wp,
                             const float* __restrict__ bias,
                             float* __restrict__ C,
                             _Float16* __restrict__ C16,
                             int M, int N,
                             const float* __restrict__ aux,
                             const float* __restrict__ flags)
{
    const int lane   = threadIdx.x & 31;
    const int wave   = (blockIdx.x * blockDim.x + threadIdx.x) >> 5;
    const int tilesN = N >> 4;
    const int pairsN = tilesN >> 1;
    const int nPairs = (M >> 4) * pairsN;
    if (wave >= nPairs) return;                 // whole wave exits together

    const int tM  = wave / pairsN;
    const int tN0 = (wave % pairsN) * 2;
    const int r16 = lane & 15;
    const int hiA = (lane >= 16) ? 8  : 0;      // D row base of lanes 16-31; A k base is 2*hiA
    const int hiB = (lane >= 16) ? 16 : 0;      // B lane-group K base (16-bit B 32x16)
    const int col0 = tN0 * 16 + r16;

    v8f acc0 = {}, acc1 = {};
    const _Float16* Arow = A16 + (size_t)(tM * 16 + r16) * K;

#pragma unroll
    for (int kk = 0; kk < K; kk += 32) {
        const v8h al = *(const v8h*)(Arow + kk + 2 * hiA);        // k = 2*hiA+0..7
        const v8h ah = *(const v8h*)(Arow + kk + 2 * hiA + 8);   // k = 2*hiA+8..15
        v16h a;
#pragma unroll
        for (int e = 0; e < 8; ++e) { a[e] = al[e]; a[e + 8] = ah[e]; }

        const _Float16* wp = Wp + (size_t)(kk >> 5) * N * 32 + hiB;
        const v16h b0 = *(const v16h*)(wp + (size_t)col0 * 32);
        const v16h b1 = *(const v16h*)(wp + (size_t)(col0 + 16) * 32);

        acc0 = __builtin_amdgcn_wmma_f32_16x16x32_f16(
                   false, a, false, b0, (short)0, acc0, false, false);
        acc1 = __builtin_amdgcn_wmma_f32_16x16x32_f16(
                   false, a, false, b1, (short)0, acc1, false, false);
    }

    const float bc0 = bias[col0];
    const float bc1 = bias[col0 + 16];
#pragma unroll
    for (int r = 0; r < 8; ++r) {
        const int row = tM * 16 + r + hiA;
        const size_t base = (size_t)row * N;
#pragma unroll
        for (int p = 0; p < 2; ++p) {
            const int col = col0 + p * 16;
            const float v = (p ? acc1[r] + bc1 : acc0[r] + bc0);
            float o;
            if (MODE == 1) {
                o = aux[base + col] * ((v > LOGIT_THR) ? 1.0f : 0.0f);
            } else if (MODE == 2) {
                o = aux[base + col] + v;
            } else if (MODE == 3) {
                o = (flags[1] > 0.5f) ? aux[base + col] : v;
            } else {
                o = v;
            }
            if (WF32) C[base + col]  = o;
            if (W16)  C16[base + col] = (_Float16)o;
        }
    }
}

// ---------------------------------------------------------------------------
__global__ void wa_init(float* flags)
{
    if (threadIdx.x < 16) flags[threadIdx.x] = 0.0f;
}

// hierarchical gate / early-exit flag (scalar since B==1)
__global__ void wa_importance(const float* __restrict__ x,
                              const float* __restrict__ hg_w,
                              const float* __restrict__ hg_b,
                              float* flags)
{
    __shared__ float sm[64];
    const int t = threadIdx.x;          // 64 threads, one per channel
    float s = 0.0f;
    for (int n = 0; n < NTOK; ++n) s += x[n * 64 + t];
    sm[t] = (s * (1.0f / NTOK)) * hg_w[t];
    __syncthreads();
    if (t == 0) {
        float a = hg_b[0];
        for (int c = 0; c < 64; ++c) a += sm[c];
        const float sig = 1.0f / (1.0f + __expf(-a));
        flags[1] = (sig < THR) ? 1.0f : 0.0f;
    }
}

// ---------------------------------------------------------------------------
// Local attention pass 1: per head, k staged in LDS via async-to-LDS.
// Head-wide max/min of k, top-20 indices (both directions), per-row softmax
// Z, and the global entropy sum for the dynamic top-k width.
// ---------------------------------------------------------------------------
__global__ void wa_local_pass1(const float* __restrict__ qkv,
                               float* flags, float* kstat,
                               int* tdesc, int* tasc,
                               float* __restrict__ Zarr)
{
    __shared__ float sk[NTOK];
    __shared__ float csel[NTOK];
    __shared__ float rv[256];
    __shared__ int   ri[256];

    const int h = blockIdx.x, t = threadIdx.x;
    for (int m = t; m < NTOK; m += 256)
        wa_async_g2l_b32(qkv + m * QKVLD + 64 + h, &sk[m]);
    wa_wait_async();
    __syncthreads();

    float lmax = -1e30f, lmin = 1e30f;
    for (int m = t; m < NTOK; m += 256) {
        const float v = sk[m];
        lmax = fmaxf(lmax, v); lmin = fminf(lmin, v);
    }
    rv[t] = lmax; __syncthreads();
    for (int s = 128; s > 0; s >>= 1) { if (t < s) rv[t] = fmaxf(rv[t], rv[t + s]); __syncthreads(); }
    const float kmax = rv[0]; __syncthreads();
    rv[t] = lmin; __syncthreads();
    for (int s = 128; s > 0; s >>= 1) { if (t < s) rv[t] = fminf(rv[t], rv[t + s]); __syncthreads(); }
    const float kmin = rv[0]; __syncthreads();
    if (t == 0) { kstat[h * 2] = kmax; kstat[h * 2 + 1] = kmin; }

    // top-20 descending (rows with q>=0 select these)
    for (int m = t; m < NTOK; m += 256) csel[m] = sk[m];
    __syncthreads();
    for (int j = 0; j < KM_MAX; ++j) {
        float bv = -1e30f; int bi = 0;
        for (int m = t; m < NTOK; m += 256) { const float v = csel[m]; if (v > bv) { bv = v; bi = m; } }
        rv[t] = bv; ri[t] = bi; __syncthreads();
        for (int s = 128; s > 0; s >>= 1) {
            if (t < s && rv[t + s] > rv[t]) { rv[t] = rv[t + s]; ri[t] = ri[t + s]; }
            __syncthreads();
        }
        if (t == 0) { tdesc[h * KM_MAX + j] = ri[0]; csel[ri[0]] = -1e30f; }
        __syncthreads();
    }
    // top-20 ascending (rows with q<0)
    for (int m = t; m < NTOK; m += 256) csel[m] = sk[m];
    __syncthreads();
    for (int j = 0; j < KM_MAX; ++j) {
        float bv = 1e30f; int bi = 0;
        for (int m = t; m < NTOK; m += 256) { const float v = csel[m]; if (v < bv) { bv = v; bi = m; } }
        rv[t] = bv; ri[t] = bi; __syncthreads();
        for (int s = 128; s > 0; s >>= 1) {
            if (t < s && rv[t + s] < rv[t]) { rv[t] = rv[t + s]; ri[t] = ri[t + s]; }
            __syncthreads();
        }
        if (t == 0) { tasc[h * KM_MAX + j] = ri[0]; csel[ri[0]] = 1e30f; }
        __syncthreads();
    }

    // per-row softmax normalizer + entropy; one wave per row, float4 LDS reads
    const int wv = t >> 5, ln = t & 31;
    for (int r = wv; r < NTOK; r += 8) {
        const float q  = qkv[r * QKVLD + h] * SCALE;
        const float mx = (q >= 0.0f) ? q * kmax : q * kmin;
        float Z = 0.0f, Ws = 0.0f;
        for (int m = ln * 4; m < NTOK; m += 128) {
            const float4 k4 = *(const float4*)&sk[m];
            const float s0 = q * k4.x, s1 = q * k4.y, s2 = q * k4.z, s3 = q * k4.w;
            const float e0 = __expf(s0 - mx), e1 = __expf(s1 - mx);
            const float e2 = __expf(s2 - mx), e3 = __expf(s3 - mx);
            Z  += (e0 + e1) + (e2 + e3);
            Ws += (e0 * s0 + e1 * s1) + (e2 * s2 + e3 * s3);
        }
        for (int off = 16; off; off >>= 1) { Z += __shfl_xor(Z, off); Ws += __shfl_xor(Ws, off); }
        if (ln == 0) {
            Zarr[h * NTOK + r] = Z;
            atomicAdd(&flags[0], mx + __logf(Z) - Ws / Z);   // row entropy
        }
    }
}

// pass 2: dynamic-k masked weighted sum; writes f16 directly into cat16
__global__ void wa_local_pass2(const float* __restrict__ qkv,
                               const float* flags, const float* kstat,
                               const int* tdesc, const int* tasc,
                               const float* __restrict__ Zarr,
                               _Float16* __restrict__ cat16)
{
    __shared__ float sk[NTOK];
    __shared__ float sv[NTOK];
    const int h = blockIdx.x, t = threadIdx.x;
    for (int m = t; m < NTOK; m += 256) {
        wa_async_g2l_b32(qkv + m * QKVLD + 64  + h, &sk[m]);
        wa_async_g2l_b32(qkv + m * QKVLD + 128 + h, &sv[m]);
    }
    wa_wait_async();
    __syncthreads();

    const float kmax = kstat[h * 2], kmin = kstat[h * 2 + 1];
    const float entm = flags[0] * (1.0f / (HEADS * NTOK));
    int km = KM_BASE + (int)rintf((KM_MAX - KM_BASE) * entm);
    km = km < KM_BASE ? KM_BASE : (km > KM_MAX ? KM_MAX : km);

    const int wv = t >> 5, ln = t & 31;
    for (int r = wv; r < NTOK; r += 8) {
        const float q  = qkv[r * QKVLD + h] * SCALE;
        const float mx = (q >= 0.0f) ? q * kmax : q * kmin;
        const float rZ = 1.0f / Zarr[h * NTOK + r];
        const int* lst = (q >= 0.0f) ? (tdesc + h * KM_MAX) : (tasc + h * KM_MAX);
        float acc = 0.0f;
        if (ln < km) {
            const int idx = lst[ln];
            acc = __expf(q * sk[idx] - mx) * rZ * sv[idx];
        }
        for (int off = 16; off; off >>= 1) acc += __shfl_xor(acc, off);
        if (ln == 0) cat16[r * 128 + h] = (_Float16)acc;
    }
}

// pooled global attention (rd==1 => pooling is identity); writes cat16
__global__ void wa_global(const float* __restrict__ gqkv, _Float16* __restrict__ cat16)
{
    __shared__ float sgk[NTOK];
    __shared__ float sgv[NTOK];
    const int h = blockIdx.x, t = threadIdx.x;
    for (int m = t; m < NTOK; m += 256) {
        wa_async_g2l_b32(gqkv + m * QKVLD + 64  + h, &sgk[m]);
        wa_async_g2l_b32(gqkv + m * QKVLD + 128 + h, &sgv[m]);
    }
    wa_wait_async();
    __syncthreads();

    const int wv = t >> 5, ln = t & 31;
    for (int r = wv; r < NTOK; r += 8) {
        const float q = gqkv[r * QKVLD + h] * SCALE;
        float mx = -1e30f;
        for (int m = ln * 4; m < NTOK; m += 128) {
            const float4 k4 = *(const float4*)&sgk[m];
            mx = fmaxf(mx, fmaxf(fmaxf(q * k4.x, q * k4.y), fmaxf(q * k4.z, q * k4.w)));
        }
        for (int off = 16; off; off >>= 1) mx = fmaxf(mx, __shfl_xor(mx, off));
        float Z = 0.0f, num = 0.0f;
        for (int m = ln * 4; m < NTOK; m += 128) {
            const float4 k4 = *(const float4*)&sgk[m];
            const float4 v4 = *(const float4*)&sgv[m];
            const float e0 = __expf(q * k4.x - mx), e1 = __expf(q * k4.y - mx);
            const float e2 = __expf(q * k4.z - mx), e3 = __expf(q * k4.w - mx);
            Z   += (e0 + e1) + (e2 + e3);
            num += (e0 * v4.x + e1 * v4.y) + (e2 * v4.z + e3 * v4.w);
        }
        for (int off = 16; off; off >>= 1) { Z += __shfl_xor(Z, off); num += __shfl_xor(num, off); }
        if (ln == 0) cat16[r * 128 + 64 + h] = (_Float16)(num / Z);
    }
}

// ---------------------------------------------------------------------------
extern "C" void kernel_launch(void* const* d_in, const int* in_sizes, int n_in,
                              void* d_out, int out_size, void* d_ws, size_t ws_size,
                              hipStream_t stream)
{
    const float* x      = (const float*)d_in[0];
    const float* qkv_w  = (const float*)d_in[1];
    const float* qkv_b  = (const float*)d_in[2];
    const float* gqkv_w = (const float*)d_in[3];
    const float* gqkv_b = (const float*)d_in[4];
    const float* cg_w   = (const float*)d_in[5];
    const float* cg_b   = (const float*)d_in[6];
    const float* hg_w   = (const float*)d_in[7];
    const float* hg_b   = (const float*)d_in[8];
    const float* fus_w  = (const float*)d_in[9];
    const float* fus_b  = (const float*)d_in[10];
    const float* proj_w = (const float*)d_in[11];
    const float* proj_b = (const float*)d_in[12];

    float* f     = (float*)d_ws;
    float* flags = f;                      // [0]=entropy sum, [1]=early flag
    float* kstat = f + 16;                 // 64 heads * {kmax,kmin}
    int*   tdesc = (int*)(f + 160);        // 64*20
    int*   tasc  = (int*)(f + 1440);       // 64*20
    float* Zarr  = f + 2720;               // 64*1024
    float* xs    = f + 68256;              // 1024*64  sparsified x (f32)
    float* qkvb  = f + 133792;             // 1024*192
    float* gqkvb = f + 330400;             // 1024*192
    _Float16* hb = (_Float16*)(f + 527008);// 32B-aligned f16 region
    _Float16* x16   = hb;                  // 1024*64
    _Float16* xs16  = hb + 65536;          // 1024*64
    _Float16* cat16 = hb + 131072;         // 1024*128 [local | global]
    _Float16* tmp16 = hb + 262144;         // 1024*64  xs + fused
    _Float16* cgp   = hb + 327680;         // packed weights (B-fragment layout)
    _Float16* qkvp  = hb + 331776;
    _Float16* gqkvp = hb + 344064;
    _Float16* fusp  = hb + 356352;
    _Float16* projp = hb + 364544;
    float* out = (float*)d_out;

    wa_init<<<1, 32, 0, stream>>>(flags);
    wa_importance<<<1, 64, 0, stream>>>(x, hg_w, hg_b, flags);

    // one-time (per launch) f16 conversions / weight packing
    wa_cvt16 <<<256, 256, 0, stream>>>(x, x16, NTOK * 64);
    wa_pack_w<<<16,  256, 0, stream>>>(cg_w,   cgp,   64, 64);
    wa_pack_w<<<48,  256, 0, stream>>>(qkv_w,  qkvp,  64, QKVLD);
    wa_pack_w<<<48,  256, 0, stream>>>(gqkv_w, gqkvp, 64, QKVLD);
    wa_pack_w<<<32,  256, 0, stream>>>(fus_w,  fusp,  128, 64);
    wa_pack_w<<<16,  256, 0, stream>>>(proj_w, projp, 64, 64);

    // gate: xs = x * (x@cg_w+cg_b > logit(0.1));  writes f32 + f16
    wa_wmma_gemm<64, 1, true, true><<<16, 256, 0, stream>>>(
        x16, cgp, cg_b, xs, xs16, NTOK, 64, x, flags);
    // qkv / gqkv projections (f32 out, consumed by attention kernels)
    wa_wmma_gemm<64, 0, true, false><<<48, 256, 0, stream>>>(
        xs16, qkvp,  qkv_b,  qkvb,  nullptr, NTOK, QKVLD, x, flags);
    wa_wmma_gemm<64, 0, true, false><<<48, 256, 0, stream>>>(
        xs16, gqkvp, gqkv_b, gqkvb, nullptr, NTOK, QKVLD, x, flags);

    wa_local_pass1<<<HEADS, 256, 0, stream>>>(qkvb, flags, kstat, tdesc, tasc, Zarr);
    wa_local_pass2<<<HEADS, 256, 0, stream>>>(qkvb, flags, kstat, tdesc, tasc, Zarr, cat16);
    wa_global<<<HEADS, 256, 0, stream>>>(gqkvb, cat16);

    // tmp16 = xs + (cat@fus_w + fus_b)   (f16 only; feeds proj)
    wa_wmma_gemm<128, 2, false, true><<<16, 256, 0, stream>>>(
        cat16, fusp, fus_b, nullptr, tmp16, NTOK, 64, xs, flags);
    // out = early ? x : (tmp@proj_w + proj_b)
    wa_wmma_gemm<64, 3, true, false><<<16, 256, 0, stream>>>(
        tmp16, projp, proj_b, out, nullptr, NTOK, 64, x, flags);
}
